// ClassForgeEnsembleGNN_67293547594216
// MI455X (gfx1250) — compile-verified, matrix operand order checked by reading, not run on hardware
//
#include <hip/hip_runtime.h>
#include <hip/hip_bf16.h>
#include <string.h>

// ---------------- problem constants ----------------
static constexpr size_t N_ = 50000;
static constexpr size_t E_ = 400000;
static constexpr size_t D_ = 128;
static constexpr size_t H_ = 2;
static constexpr size_t R_ = 5;
static constexpr int    OUTD = 3 * 128;          // 384, output row stride
static constexpr float  NEG_SLOPE = 0.2f;
static constexpr int    RPW = 5;                 // row tiles per wave (3125 = 625*5)

// ---------------- workspace layout (bytes) ----------------
static constexpr size_t al256(size_t x) { return (x + 255) & ~(size_t)255; }
static constexpr size_t SZ_BF_ND = N_ * D_ * 2;                      // bf16 [N,D]
static constexpr size_t O_XBF   = 0;
static constexpr size_t O_X1BF  = al256(O_XBF  + SZ_BF_ND);
static constexpr size_t O_X2BF  = al256(O_X1BF + SZ_BF_ND);
static constexpr size_t O_AGGBF = al256(O_X2BF + SZ_BF_ND);
static constexpr size_t O_WL    = al256(O_AGGBF + SZ_BF_ND);         // packed [128x128]
static constexpr size_t O_WR    = al256(O_WL + D_ * D_ * 2);
static constexpr size_t O_WGAT  = al256(O_WR + D_ * D_ * 2);         // packed [128x256]
static constexpr size_t O_WCAT  = al256(O_WGAT + D_ * H_ * D_ * 2);  // packed [128x768]
static constexpr size_t O_WTMP  = al256(O_WCAT + D_ * (D_ + R_ * D_) * 2); // unpacked bf16 temp
static constexpr size_t O_CNT   = al256(O_WTMP + D_ * (D_ + R_ * D_) * 2); // [N] f32
static constexpr size_t O_AS    = al256(O_CNT + N_ * 4);             // [N,H] f32
static constexpr size_t O_AD    = al256(O_AS + N_ * H_ * 4);
static constexpr size_t O_MENC  = al256(O_AD + N_ * H_ * 4);         // [N,H] u32 (monotone enc)
static constexpr size_t O_DEN   = al256(O_MENC + N_ * H_ * 4);
static constexpr size_t O_EX    = al256(O_DEN + N_ * H_ * 4);        // [(E+N),H] f32
static constexpr size_t O_CREL  = al256(O_EX + (E_ + N_) * H_ * 4);  // [N,R] f32
static constexpr size_t O_ACC   = al256(O_CREL + N_ * R_ * 4);       // [N,D] f32: agg_sum then x2_pre
static constexpr size_t O_H     = al256(O_ACC + N_ * D_ * 4);        // [N,H*D] f32
static constexpr size_t O_XR    = al256(O_H + N_ * H_ * D_ * 4);     // [R,N,D] bf16
static constexpr size_t WS_NEED = O_XR + R_ * N_ * D_ * 2;

// ---------------- types ----------------
typedef __attribute__((ext_vector_type(16))) __bf16       v16bf;
typedef __attribute__((ext_vector_type(8)))  float        v8f;
typedef __attribute__((ext_vector_type(8)))  unsigned int v8u;

// ---------------- device helpers ----------------
__device__ __forceinline__ unsigned short f2bf(float f) {
    unsigned u = __float_as_uint(f);
    unsigned r = u + 0x7FFFu + ((u >> 16) & 1u);     // round-to-nearest-even
    return (unsigned short)(r >> 16);
}
__device__ __forceinline__ float bf2f(unsigned short h) {
    return __uint_as_float(((unsigned)h) << 16);
}
// monotone float<->uint mapping for atomicMax-based segment max
__device__ __forceinline__ unsigned fenc(float f) {
    unsigned u = __float_as_uint(f);
    return (u & 0x80000000u) ? ~u : (u | 0x80000000u);
}
__device__ __forceinline__ float fdec(unsigned e) {
    unsigned u = (e & 0x80000000u) ? (e ^ 0x80000000u) : ~e;
    return __uint_as_float(u);
}
// fragment-layout k offsets: VGPR i holds pair at kh + {0,2,4,6}[i] (i<4) or kh + 16 + {0,2,4,6}[i-4]
__device__ __forceinline__ int frag_koff(int i) { return (i < 4) ? (2 * i) : (16 + 2 * (i - 4)); }

// packed-B fragment read: lane's 8 uints are contiguous -> 2x b128 loads
__device__ __forceinline__ v16bf load_b_packed(const unsigned* Bp, int ncols16,
                                               int kt, int ct, int lane) {
    const v8u* p = (const v8u*)(Bp + (((size_t)kt * ncols16 + ct) * 32 + lane) * 8);
    return __builtin_bit_cast(v16bf, *p);
}

__device__ __forceinline__ v8f wmma_bf16(v16bf a, v16bf b, v8f c) {
    return __builtin_amdgcn_wmma_f32_16x16x32_bf16(false, a, false, b, (short)0, c, false, false);
}

// ---------------- utility kernels ----------------
__global__ void fill_f32_k(float* p, float v, int n) {
    int i = blockIdx.x * blockDim.x + threadIdx.x;
    if (i < n) p[i] = v;
}
__global__ void fill_u32_k(unsigned* p, unsigned v, int n) {
    int i = blockIdx.x * blockDim.x + threadIdx.x;
    if (i < n) p[i] = v;
}
__global__ void cvt_bf16_k(const float* s, unsigned short* d, int n) {
    int i = blockIdx.x * blockDim.x + threadIdx.x;
    if (i < n) d[i] = f2bf(s[i]);
}
// pack W_root | W_rgcn[0..4] into one [128 x 768] bf16 matrix (unpacked row-major)
__global__ void build_wcat_k(const float* Wroot, const float* Wrgcn, unsigned short* Wcat) {
    int i = blockIdx.x * blockDim.x + threadIdx.x;          // D*768
    if (i >= (int)(D_ * 768)) return;
    int d = i / 768, j = i % 768;
    float v = (j < 128) ? Wroot[d * 128 + j]
                        : Wrgcn[(size_t)((j - 128) >> 7) * D_ * D_ + (size_t)d * 128 + (j & 127)];
    Wcat[i] = f2bf(v);
}
// repack row-major bf16 [128 x ncols] into WMMA B-fragment layout:
// uint index = ((kt*ncols16 + ct)*32 + lane)*8 + i
__global__ void pack_b_k(const unsigned short* Bu, int ncols, unsigned* Bp, int total) {
    int i = blockIdx.x * blockDim.x + threadIdx.x;          // total = 64*ncols
    if (i >= total) return;
    const int ncols16 = ncols >> 4;
    int ii = i & 7, lane = (i >> 3) & 31, rest = i >> 8;
    int ct = rest % ncols16, kt = rest / ncols16;
    int n = ct * 16 + (lane & 15);
    int k = kt * 32 + ((lane >> 4) << 3) + frag_koff(ii);
    unsigned lo = Bu[(size_t)k * ncols + n];
    unsigned hi = Bu[(size_t)(k + 1) * ncols + n];
    Bp[i] = lo | (hi << 16);
}

// ---------------- SAGE ----------------
__global__ void sage_scatter_k(const float* x, const int* src, const int* dst,
                               float* agg_sum, float* cnt) {
    int i = blockIdx.x * blockDim.x + threadIdx.x;          // E*D
    if (i >= (int)(E_ * D_)) return;
    int e = i >> 7, d = i & 127;
    int s = src[e], t = dst[e];
    atomicAdd(&agg_sum[(size_t)t * 128 + d], x[(size_t)s * 128 + d]);
    if (d == 0) atomicAdd(&cnt[t], 1.0f);
}
__global__ void sage_finalize_k(const float* agg_sum, const float* cnt, unsigned short* agg_bf) {
    int i = blockIdx.x * blockDim.x + threadIdx.x;          // N*D
    if (i >= (int)(N_ * D_)) return;
    int n = i >> 7;
    agg_bf[i] = f2bf(agg_sum[i] / fmaxf(cnt[n], 1.0f));
}

// ---------------- WMMA GEMMs ----------------
// Cooperative LDS fill of fragment-packed A: 5 row tiles x 4 k tiles, 5120 uints (20KB).
// Packed pair (k,k+1) is contiguous in row-major A -> plain u32 global load.
__device__ __forceinline__ void fill_a_lds(unsigned* Alds, const unsigned short* A,
                                           int row0, int tid) {
#pragma unroll 4
    for (int p = tid; p < RPW * 4 * 256; p += 256) {
        int ii = p & 7, lane = (p >> 3) & 31, jkt = p >> 8;
        int kt = jkt & 3, j = jkt >> 2;
        int m = row0 + j * 16 + (lane & 15);
        int k = kt * 32 + ((lane >> 4) << 3) + frag_koff(ii);
        Alds[p] = *(const unsigned*)(A + (size_t)m * 128 + k);
    }
}
__device__ __forceinline__ v16bf read_a_lds(const unsigned* Alds, int j, int kt, int lane) {
    const v8u* p = (const v8u*)(Alds + (((size_t)j * 4 + kt) * 32 + lane) * 8);
    return __builtin_bit_cast(v16bf, *p);
}

// x1 = relu(agg@Wl + x@Wr + b); writes d_out cols [0,128) and x1 bf16
__global__ __launch_bounds__(256) void gemm_x1_k(const unsigned short* Agg, const unsigned short* X,
                                                 const unsigned* WlP, const unsigned* WrP,
                                                 const float* bias, float* out, unsigned short* x1bf) {
    __shared__ unsigned AldsA[RPW * 4 * 256];
    __shared__ unsigned AldsX[RPW * 4 * 256];
    const int tid = threadIdx.x;
    const int row0 = blockIdx.x * (16 * RPW);
    fill_a_lds(AldsA, Agg, row0, tid);
    fill_a_lds(AldsX, X, row0, tid);
    __syncthreads();
    const int lane = tid & 31, wave = tid >> 5;             // wave = col tile (8 x 16 = 128 cols)
    v8f acc[RPW] = {};
#pragma unroll
    for (int kt = 0; kt < 4; ++kt) {
        v16bf bl = load_b_packed(WlP, 8, kt, wave, lane);
        v16bf br = load_b_packed(WrP, 8, kt, wave, lane);
#pragma unroll
        for (int j = 0; j < RPW; ++j) {
            acc[j] = wmma_bf16(read_a_lds(AldsA, j, kt, lane), bl, acc[j]);
            acc[j] = wmma_bf16(read_a_lds(AldsX, j, kt, lane), br, acc[j]);
        }
    }
    const int n = wave * 16 + (lane & 15);
    const float bv = bias[n];
#pragma unroll
    for (int j = 0; j < RPW; ++j) {
        const int mbase = row0 + j * 16 + ((lane >> 4) << 3);
#pragma unroll
        for (int i = 0; i < 8; ++i) {
            const float v = fmaxf(acc[j][i] + bv, 0.0f);
            out[(size_t)(mbase + i) * OUTD + n] = v;
            x1bf[(size_t)(mbase + i) * 128 + n] = f2bf(v);
        }
    }
}
// h = x1 @ Wgat  ([N,256] f32)
__global__ __launch_bounds__(256) void gemm_h_k(const unsigned short* X1, const unsigned* WgP,
                                                float* h) {
    __shared__ unsigned Alds[RPW * 4 * 256];
    const int tid = threadIdx.x;
    const int row0 = blockIdx.x * (16 * RPW);
    fill_a_lds(Alds, X1, row0, tid);
    __syncthreads();
    const int lane = tid & 31;
    const int ct = blockIdx.y * 8 + (tid >> 5);             // 16 col tiles total
    v8f acc[RPW] = {};
#pragma unroll
    for (int kt = 0; kt < 4; ++kt) {
        v16bf b = load_b_packed(WgP, 16, kt, ct, lane);
#pragma unroll
        for (int j = 0; j < RPW; ++j)
            acc[j] = wmma_bf16(read_a_lds(Alds, j, kt, lane), b, acc[j]);
    }
    const int n = ct * 16 + (lane & 15);
#pragma unroll
    for (int j = 0; j < RPW; ++j) {
        const int mbase = row0 + j * 16 + ((lane >> 4) << 3);
#pragma unroll
        for (int i = 0; i < 8; ++i)
            h[(size_t)(mbase + i) * 256 + n] = acc[j][i];
    }
}
// x2 @ [Wroot|Wrgcn] ([N,768]): col<128 -> d_out x3-init (+bias); else -> xr bf16
__global__ __launch_bounds__(256) void gemm_rgcn_k(const unsigned short* X2, const unsigned* WcP,
                                                   const float* b_rgcn, float* out,
                                                   unsigned short* xr) {
    __shared__ unsigned Alds[RPW * 4 * 256];
    const int tid = threadIdx.x;
    const int row0 = blockIdx.x * (16 * RPW);
    fill_a_lds(Alds, X2, row0, tid);
    __syncthreads();
    const int lane = tid & 31;
    const int ct = blockIdx.y * 8 + (tid >> 5);             // 48 col tiles total
    v8f acc[RPW] = {};
#pragma unroll
    for (int kt = 0; kt < 4; ++kt) {
        v16bf b = load_b_packed(WcP, 48, kt, ct, lane);
#pragma unroll
        for (int j = 0; j < RPW; ++j)
            acc[j] = wmma_bf16(read_a_lds(Alds, j, kt, lane), b, acc[j]);
    }
    const int n = ct * 16 + (lane & 15);
    if (n < 128) {
        const float bv = b_rgcn[n];
#pragma unroll
        for (int j = 0; j < RPW; ++j) {
            const int mbase = row0 + j * 16 + ((lane >> 4) << 3);
#pragma unroll
            for (int i = 0; i < 8; ++i)
                out[(size_t)(mbase + i) * OUTD + 256 + n] = acc[j][i] + bv;
        }
    } else {
        const int r = (n >> 7) - 1, c = n & 127;
#pragma unroll
        for (int j = 0; j < RPW; ++j) {
            const int mbase = row0 + j * 16 + ((lane >> 4) << 3);
#pragma unroll
            for (int i = 0; i < 8; ++i)
                xr[((size_t)r * N_ + (mbase + i)) * 128 + c] = f2bf(acc[j][i]);
        }
    }
}

// ---------------- GAT ----------------
__global__ void gat_scores_k(const float* h, const float* att_src, const float* att_dst,
                             float* a_s, float* a_d) {
    const int lane = threadIdx.x & 31;
    const int wid = blockIdx.x * 8 + (threadIdx.x >> 5);    // (n,hh) pair per wave
    if (wid >= (int)(N_ * H_)) return;
    const int n = wid >> 1, hh = wid & 1;
    const float* hp = h + (size_t)n * 256 + hh * 128;
    const float* as = att_src + hh * 128;
    const float* ad = att_dst + hh * 128;
    float ss = 0.f, sd = 0.f;
#pragma unroll
    for (int c = lane; c < 128; c += 32) { float hv = hp[c]; ss += hv * as[c]; sd += hv * ad[c]; }
#pragma unroll
    for (int off = 16; off > 0; off >>= 1) {
        ss += __shfl_down(ss, off, 32);
        sd += __shfl_down(sd, off, 32);
    }
    if (lane == 0) { a_s[wid] = ss; a_d[wid] = sd; }
}
__device__ __forceinline__ void edge_st(int i, const int* src, const int* dst, int& s, int& t) {
    if (i < (int)E_) { s = src[i]; t = dst[i]; } else { s = t = i - (int)E_; }  // self-loops appended
}
__global__ void gat_max_k(const int* src, const int* dst, const float* a_s, const float* a_d,
                          unsigned* menc) {
    int i = blockIdx.x * blockDim.x + threadIdx.x;          // E+N
    if (i >= (int)(E_ + N_)) return;
    int s, t; edge_st(i, src, dst, s, t);
#pragma unroll
    for (int hh = 0; hh < 2; ++hh) {
        float e = a_s[s * 2 + hh] + a_d[t * 2 + hh];
        e = (e > 0.f) ? e : e * NEG_SLOPE;
        atomicMax(&menc[t * 2 + hh], fenc(e));
    }
}
__global__ void gat_denom_k(const int* src, const int* dst, const float* a_s, const float* a_d,
                            const unsigned* menc, float* denom, float* ex_buf) {
    int i = blockIdx.x * blockDim.x + threadIdx.x;          // E+N
    if (i >= (int)(E_ + N_)) return;
    int s, t; edge_st(i, src, dst, s, t);
#pragma unroll
    for (int hh = 0; hh < 2; ++hh) {
        float e = a_s[s * 2 + hh] + a_d[t * 2 + hh];
        e = (e > 0.f) ? e : e * NEG_SLOPE;
        float ex = expf(e - fdec(menc[t * 2 + hh]));
        ex_buf[(size_t)i * 2 + hh] = ex;
        atomicAdd(&denom[t * 2 + hh], ex);
    }
}
__global__ void gat_scatter_k(const int* src, const int* dst, const float* h,
                              const float* ex_buf, const float* denom, float* x2pre) {
    int i = blockIdx.x * blockDim.x + threadIdx.x;          // (E+N)*D
    if (i >= (int)((E_ + N_) * D_)) return;
    int idx = i >> 7, d = i & 127;
    int s, t; edge_st(idx, src, dst, s, t);
    const float a0 = ex_buf[(size_t)idx * 2 + 0] / denom[t * 2 + 0];
    const float a1 = ex_buf[(size_t)idx * 2 + 1] / denom[t * 2 + 1];
    const float msg = 0.5f * (h[(size_t)s * 256 + d] * a0 + h[(size_t)s * 256 + 128 + d] * a1);
    atomicAdd(&x2pre[(size_t)t * 128 + d], msg);
}
__global__ void gat_finalize_k(const float* x2pre, const float* b_gat,
                               float* out, unsigned short* x2bf) {
    int i = blockIdx.x * blockDim.x + threadIdx.x;          // N*D
    if (i >= (int)(N_ * D_)) return;
    int n = i >> 7, d = i & 127;
    float v = fmaxf(x2pre[i] + b_gat[d], 0.0f);
    out[(size_t)n * OUTD + 128 + d] = v;
    x2bf[i] = f2bf(v);
}

// ---------------- RGCN sparse ----------------
__global__ void rgcn_count_k(const int* dst, const int* etype, float* c_rel) {
    int e = blockIdx.x * blockDim.x + threadIdx.x;
    if (e >= (int)E_) return;
    atomicAdd(&c_rel[dst[e] * 5 + etype[e]], 1.0f);
}
__global__ void rgcn_scatter_k(const int* src, const int* dst, const int* etype,
                               const unsigned short* xr, const float* c_rel, float* out) {
    int i = blockIdx.x * blockDim.x + threadIdx.x;          // E*D
    if (i >= (int)(E_ * D_)) return;
    int e = i >> 7, d = i & 127;
    int s = src[e], t = dst[e], r = etype[e];
    float c = fmaxf(c_rel[t * 5 + r], 1.0f);
    float v = bf2f(xr[((size_t)r * N_ + s) * 128 + d]) / c;
    atomicAdd(&out[(size_t)t * OUTD + 256 + d], v);
}

// ---------------- launcher ----------------
static inline unsigned gdiv(size_t a, unsigned b) { return (unsigned)((a + b - 1) / b); }

extern "C" void kernel_launch(void* const* d_in, const int* in_sizes, int n_in,
                              void* d_out, int out_size, void* d_ws, size_t ws_size,
                              hipStream_t stream) {
    (void)in_sizes; (void)n_in; (void)out_size;
    if (ws_size < WS_NEED) return;

    const float* x        = (const float*)d_in[0];
    const int*   eidx     = (const int*)d_in[1];
    const int*   etype    = (const int*)d_in[2];
    const float* W_sage_l = (const float*)d_in[3];
    const float* b_sage   = (const float*)d_in[4];
    const float* W_sage_r = (const float*)d_in[5];
    const float* W_gat    = (const float*)d_in[6];
    const float* att_src  = (const float*)d_in[7];
    const float* att_dst  = (const float*)d_in[8];
    const float* b_gat    = (const float*)d_in[9];
    const float* W_rgcn   = (const float*)d_in[10];
    const float* W_root   = (const float*)d_in[11];
    const float* b_rgcn   = (const float*)d_in[12];

    const int* src = eidx;
    const int* dst = eidx + E_;
    float* out = (float*)d_out;

    char* ws = (char*)d_ws;
    unsigned short* xbf   = (unsigned short*)(ws + O_XBF);
    unsigned short* x1bf  = (unsigned short*)(ws + O_X1BF);
    unsigned short* x2bf  = (unsigned short*)(ws + O_X2BF);
    unsigned short* aggbf = (unsigned short*)(ws + O_AGGBF);
    unsigned* wlP   = (unsigned*)(ws + O_WL);
    unsigned* wrP   = (unsigned*)(ws + O_WR);
    unsigned* wgatP = (unsigned*)(ws + O_WGAT);
    unsigned* wcatP = (unsigned*)(ws + O_WCAT);
    unsigned short* wtmp = (unsigned short*)(ws + O_WTMP);
    float*    cnt   = (float*)(ws + O_CNT);
    float*    a_s   = (float*)(ws + O_AS);
    float*    a_d   = (float*)(ws + O_AD);
    unsigned* menc  = (unsigned*)(ws + O_MENC);
    float*    denom = (float*)(ws + O_DEN);
    float*    exb   = (float*)(ws + O_EX);
    float*    crel  = (float*)(ws + O_CREL);
    float*    accf  = (float*)(ws + O_ACC);     // agg_sum, later x2_pre (aliased)
    float*    hbuf  = (float*)(ws + O_H);
    unsigned short* xr = (unsigned short*)(ws + O_XR);

    const int B = 256;
    float ninf = -3.0e38f; unsigned ubits; memcpy(&ubits, &ninf, 4);
    const unsigned ENC_NINF = (ubits & 0x80000000u) ? ~ubits : (ubits | 0x80000000u);
    const unsigned GEMM_GX = (unsigned)(N_ / (16 * RPW));   // 625

    // --- init + precision conversion + weight fragment packing ---
    fill_f32_k<<<gdiv(N_ * D_, B), B, 0, stream>>>(accf, 0.f, (int)(N_ * D_));       // agg_sum
    fill_f32_k<<<gdiv(N_, B), B, 0, stream>>>(cnt, 0.f, (int)N_);
    fill_f32_k<<<gdiv(N_ * H_, B), B, 0, stream>>>(denom, 0.f, (int)(N_ * H_));
    fill_u32_k<<<gdiv(N_ * H_, B), B, 0, stream>>>(menc, ENC_NINF, (int)(N_ * H_));
    fill_f32_k<<<gdiv(N_ * R_, B), B, 0, stream>>>(crel, 0.f, (int)(N_ * R_));
    cvt_bf16_k<<<gdiv(N_ * D_, B), B, 0, stream>>>(x, xbf, (int)(N_ * D_));
    // sequential reuse of wtmp is safe: all launches are stream-ordered
    cvt_bf16_k<<<gdiv(D_ * D_, B), B, 0, stream>>>(W_sage_l, wtmp, (int)(D_ * D_));
    pack_b_k<<<gdiv(64 * 128, B), B, 0, stream>>>(wtmp, 128, wlP, 64 * 128);
    cvt_bf16_k<<<gdiv(D_ * D_, B), B, 0, stream>>>(W_sage_r, wtmp, (int)(D_ * D_));
    pack_b_k<<<gdiv(64 * 128, B), B, 0, stream>>>(wtmp, 128, wrP, 64 * 128);
    cvt_bf16_k<<<gdiv(D_ * H_ * D_, B), B, 0, stream>>>(W_gat, wtmp, (int)(D_ * H_ * D_));
    pack_b_k<<<gdiv(64 * 256, B), B, 0, stream>>>(wtmp, 256, wgatP, 64 * 256);
    build_wcat_k<<<gdiv(D_ * 768, B), B, 0, stream>>>(W_root, W_rgcn, wtmp);
    pack_b_k<<<gdiv(64 * 768, B), B, 0, stream>>>(wtmp, 768, wcatP, 64 * 768);

    // --- SAGE ---
    sage_scatter_k<<<gdiv(E_ * D_, B), B, 0, stream>>>(x, src, dst, accf, cnt);
    sage_finalize_k<<<gdiv(N_ * D_, B), B, 0, stream>>>(accf, cnt, aggbf);
    fill_f32_k<<<gdiv(N_ * D_, B), B, 0, stream>>>(accf, 0.f, (int)(N_ * D_));       // now x2_pre
    gemm_x1_k<<<GEMM_GX, B, 0, stream>>>(aggbf, xbf, wlP, wrP, b_sage, out, x1bf);

    // --- GAT ---
    gemm_h_k<<<dim3(GEMM_GX, 2), B, 0, stream>>>(x1bf, wgatP, hbuf);
    gat_scores_k<<<gdiv(N_ * H_, 8), B, 0, stream>>>(hbuf, att_src, att_dst, a_s, a_d);
    gat_max_k<<<gdiv(E_ + N_, B), B, 0, stream>>>(src, dst, a_s, a_d, menc);
    gat_denom_k<<<gdiv(E_ + N_, B), B, 0, stream>>>(src, dst, a_s, a_d, menc, denom, exb);
    gat_scatter_k<<<gdiv((E_ + N_) * D_, B), B, 0, stream>>>(src, dst, hbuf, exb, denom, accf);
    gat_finalize_k<<<gdiv(N_ * D_, B), B, 0, stream>>>(accf, b_gat, out, x2bf);

    // --- RGCN ---
    rgcn_count_k<<<gdiv(E_, B), B, 0, stream>>>(dst, etype, crel);
    gemm_rgcn_k<<<dim3(GEMM_GX, 6), B, 0, stream>>>(x2bf, wcatP, b_rgcn, out, xr);
    rgcn_scatter_k<<<gdiv(E_ * D_, B), B, 0, stream>>>(src, dst, etype, xr, crel, out);
}
